// MultiHeadAttention_77060303225538
// MI455X (gfx1250) — compile-verified, matrix operand order checked by reading, not run on hardware
//
#include <hip/hip_runtime.h>
#include <hip/hip_bf16.h>

// ---------------------------------------------------------------------------
// MHA for MI455X (gfx1250, wave32, WMMA 16x16x32 bf16, f32 accumulate)
// B=4, TQ=TKV=2048, D_MODEL=1024, H=16, DQ=DK=DV=64
// ---------------------------------------------------------------------------

#define BATCH 4
#define TQN   2048
#define TKVN  2048
#define DM    1024
#define NH    16
#define HD    64
#define MROWS (BATCH * TQN)   // 8192
#define LDSP  17              // padded LDS row stride (dwords) -> conflict-free

typedef __attribute__((ext_vector_type(16))) __bf16 v16bf;
typedef __attribute__((ext_vector_type(8)))  float  v8f;
typedef __attribute__((ext_vector_type(4)))  unsigned int u32x4;
typedef __attribute__((ext_vector_type(8)))  int i32x8;
typedef __attribute__((ext_vector_type(4)))  int i32x4;

#if defined(__has_builtin)
#if __has_builtin(__builtin_amdgcn_tensor_load_to_lds) && \
    __has_builtin(__builtin_amdgcn_s_wait_tensorcnt)
#define HAVE_TDM 1
#endif
#endif

#ifndef HAVE_TDM
#warning "CDNA5 TDM builtins not found: flash kernel uses cooperative LDS staging fallback"
#endif

union FragB { v16bf v; unsigned int u[8]; };

__device__ __forceinline__ unsigned short f2bf(float f) {
  unsigned int u = __float_as_uint(f);
  unsigned int r = u + 0x7FFFu + ((u >> 16) & 1u);   // round-to-nearest-even
  return (unsigned short)(r >> 16);
}
__device__ __forceinline__ unsigned int pack2bf(float a, float b) {
  return (unsigned int)f2bf(a) | ((unsigned int)f2bf(b) << 16);
}
// K-pair index inside a 16-bit WMMA fragment (ISA 7.12.2): vgpr v holds K-pair
// (v<4 ? v : 4+v) + (lane>=16 ? 4 : 0)   [in units of bf16 pairs]
__device__ __forceinline__ int pairIdx(int v, int hl) {
  return (v < 4 ? v : 4 + v) + hl * 4;
}

#ifdef HAVE_TDM
// Issue a TDM 2D-tile load: tile (tile0 x tile1) of 2-byte elements from a
// row-major tensor (row stride stride0 elements) into LDS, with +1 dword of
// LDS padding per tile row (pad_interval selects the row length in dwords).
__device__ __forceinline__ void tdm_load2d(unsigned long long gaddr,
                                           unsigned ldsaddr, unsigned td0,
                                           unsigned td1, unsigned tile0,
                                           unsigned tile1, unsigned stride0,
                                           unsigned padint) {
  u32x4 g0;
  g0.x = 1u;                                        // count=1, user mode
  g0.y = ldsaddr;                                   // LDS byte address
  g0.z = (unsigned)(gaddr & 0xFFFFFFFFull);         // global_addr[31:0]
  g0.w = (unsigned)((gaddr >> 32) & 0x01FFFFFFull)  // global_addr[56:32]
         | 0x80000000u;                             // type=2 ("image")
  i32x8 g1;
  g1[0] = (int)((1u << 16) |        // data_size = 2 bytes
                (1u << 20) |        // pad_enable
                (padint << 22));    // pad_interval; pad_amount=0 -> 1 dword
  g1[1] = (int)((td0 & 0xFFFFu) << 16);                 // tensor_dim0[15:0]
  g1[2] = (int)((td0 >> 16) | ((td1 & 0xFFFFu) << 16)); // td0 hi | td1 lo
  g1[3] = (int)((td1 >> 16) | (tile0 << 16));           // td1 hi | tile_dim0
  g1[4] = (int)tile1;                                   // tile_dim1, tile_dim2=0
  g1[5] = (int)stride0;                                 // tensor_dim0_stride lo
  g1[6] = 0;
  g1[7] = 0;
  i32x4 z4 = {};
#if __clang_major__ >= 23
  i32x8 z8 = {};
  __builtin_amdgcn_tensor_load_to_lds(g0, g1, z4, z4, z8, 0);
#else
  __builtin_amdgcn_tensor_load_to_lds(g0, g1, z4, z4, 0);
#endif
}
#endif  // HAVE_TDM

// ---------------------------------------------------------------------------
// GEMM: C[M,N] = A[M,K] * W[K,N] + bias   (K=1024 here)
// Block: 256 threads (8 waves), 128x64 output tile, K-step 32.
// Software pipelined: global loads for tile k+1 are issued right after the
// barrier so their latency overlaps the LDS fragment reads + 4 WMMAs of tile k.
// Wave w: rows (w&3)*32..+32, cols (w>>2)*32..+32  (2x2 WMMA tiles, 4 WMMA).
// OUTMODE 0: bf16 row-major out0 (ld=N)
// OUTMODE 1: split: col<1024 -> K bf16 row-major (ld=1024); col>=1024 -> V
//            transposed bf16: Vt[((b*NH+h)*HD+dv)*TKV + t]
// OUTMODE 2: f32 row-major out0 (ld=N)
// ---------------------------------------------------------------------------
__device__ __forceinline__ void storeKV(void* kbuf, void* vtbuf, int row, int col,
                                        float val) {
  unsigned short bf = f2bf(val);
  if (col < NH * HD) {
    ((unsigned short*)kbuf)[(size_t)row * (NH * HD) + col] = bf;
  } else {
    int n2 = col - NH * HD;
    int h = n2 >> 6, dv = n2 & 63;
    int b = row >> 11, t = row & (TKVN - 1);
    ((unsigned short*)vtbuf)[(size_t)(((b * NH + h) << 6) + dv) * TKVN + t] = bf;
  }
}

template <bool A_BF16, int OUTMODE>
__global__ __launch_bounds__(256) void mha_gemm_kernel(
    const void* __restrict__ Ap, const float* __restrict__ W,
    const float* __restrict__ bias, void* __restrict__ out0,
    void* __restrict__ out1, int M, int K, int N) {
  __shared__ unsigned int ldsA[128 * LDSP];  // [row][kpair] packed bf16x2
  __shared__ unsigned int ldsB[64 * LDSP];   // [col][kpair] packed bf16x2

  const int tid = threadIdx.x;
  const int lane = tid & 31;
  const int w = tid >> 5;
  const int wm = w & 3, wn = w >> 2;
  const int hl = lane >> 4;
  const int l16 = lane & 15;
  const int rowbase = blockIdx.y * 128;
  const int colbase = blockIdx.x * 64;

  v8f acc00 = {}, acc01 = {}, acc10 = {}, acc11 = {};

  unsigned int ga[8], gb[4];
  auto load_tile = [&](int k0) {
#pragma unroll
    for (int i = 0; i < 8; ++i) {
      int e = tid + i * 256;  // 0..2047
      int ca = e & 15, ra = e >> 4;
      if (A_BF16) {
        const unsigned int* A32 = (const unsigned int*)Ap;
        ga[i] = A32[((size_t)(rowbase + ra) * K + k0) / 2 + ca];
      } else {
        const float* p =
            (const float*)Ap + (size_t)(rowbase + ra) * K + k0 + ca * 2;
        ga[i] = pack2bf(p[0], p[1]);
      }
      if (i < 4) {
        int nb = e & 63, cb = e >> 6;
        const float* q = W + (size_t)(k0 + cb * 2) * N + colbase + nb;
        gb[i] = pack2bf(q[0], q[N]);
      }
    }
  };

  load_tile(0);

  for (int k0 = 0; k0 < K; k0 += 32) {
    // stage registers -> LDS
#pragma unroll
    for (int i = 0; i < 8; ++i) {
      int e = tid + i * 256;
      ldsA[(e >> 4) * LDSP + (e & 15)] = ga[i];
      if (i < 4) ldsB[(e & 63) * LDSP + (e >> 6)] = gb[i];
    }
    __syncthreads();
    // issue next tile's global loads early (overlaps with WMMA below)
    if (k0 + 32 < K) load_tile(k0 + 32);

    FragB a0, a1, b0, b1;
    const int arow = wm * 32 + l16;
    const int bn0 = wn * 32 + l16;
#pragma unroll
    for (int v = 0; v < 8; ++v) {
      int pi = pairIdx(v, hl);
      a0.u[v] = ldsA[arow * LDSP + pi];
      a1.u[v] = ldsA[(arow + 16) * LDSP + pi];
      b0.u[v] = ldsB[bn0 * LDSP + pi];
      b1.u[v] = ldsB[(bn0 + 16) * LDSP + pi];
    }
    acc00 = __builtin_amdgcn_wmma_f32_16x16x32_bf16(false, a0.v, false, b0.v,
                                                    (short)0, acc00, false, false);
    acc01 = __builtin_amdgcn_wmma_f32_16x16x32_bf16(false, a0.v, false, b1.v,
                                                    (short)0, acc01, false, false);
    acc10 = __builtin_amdgcn_wmma_f32_16x16x32_bf16(false, a1.v, false, b0.v,
                                                    (short)0, acc10, false, false);
    acc11 = __builtin_amdgcn_wmma_f32_16x16x32_bf16(false, a1.v, false, b1.v,
                                                    (short)0, acc11, false, false);
    __syncthreads();
  }

  // epilogue: C layout -> row = base + mt*16 + v + 8*hl, col = base + nt*16 + l16
  const int row0 = rowbase + wm * 32;
  const int col0 = colbase + wn * 32 + l16;
  const float bias0 = bias[col0];
  const float bias1 = bias[col0 + 16];
  const v8f* accs[4] = {&acc00, &acc01, &acc10, &acc11};
#pragma unroll
  for (int mt = 0; mt < 2; ++mt) {
#pragma unroll
    for (int nt = 0; nt < 2; ++nt) {
      const v8f& A = *accs[mt * 2 + nt];
      const int col = col0 + nt * 16;
      const float bs = nt ? bias1 : bias0;
#pragma unroll
      for (int v = 0; v < 8; ++v) {
        int rv = row0 + mt * 16 + v + 8 * hl;
        float o = A[v] + bs;
        if (OUTMODE == 0) {
          ((unsigned short*)out0)[(size_t)rv * N + col] = f2bf(o);
        } else if (OUTMODE == 1) {
          storeKV(out0, out1, rv, col, o);
        } else {
          ((float*)out0)[(size_t)rv * N + col] = o;
        }
      }
    }
  }
}

// ---------------------------------------------------------------------------
// Flash attention: grid = B*H*(TQ/128), block = 256 (8 waves).
// All 8 waves share one (b,h). K/V tiles are staged once per block into
// double-buffered LDS via the Tensor Data Mover (issue tile j+1 before
// computing tile j; s_wait_tensorcnt<=2 retires tile j). Cooperative-load
// fallback when the TDM builtin is unavailable.
// Wave handles 16 q rows; KV steps of 32; online softmax.
// Q,K bf16 row-major (B*T,1024); V transposed bf16 [((b*NH+h)*HD+dv)][TKV].
// Output bf16 row-major (B*TQ, 1024).
// ---------------------------------------------------------------------------
__global__ __launch_bounds__(256) void mha_flash_kernel(
    const unsigned short* __restrict__ Qb, const unsigned short* __restrict__ Kb,
    const unsigned short* __restrict__ Vt,
    const unsigned char* __restrict__ padmask,
    const int* __restrict__ causal_flag, unsigned short* __restrict__ Ob) {
  __shared__ unsigned int ldsK[2][32 * 33];  // 32 kv rows x (32+1 pad) dwords
  __shared__ unsigned int ldsV[2][64 * 17];  // 64 dv rows x (16+1 pad) dwords
  __shared__ unsigned int pLds[8][16 * 16];  // per-wave 16x32 bf16 P staging

  const int tid = threadIdx.x;
  const int lane = tid & 31;
  const int w = tid >> 5;
  const int hl = lane >> 4;
  const int l16 = lane & 15;

  const int blk = blockIdx.x;
  const int qblk = blk & (TQN / 128 - 1);         // 16 q-blocks
  const int h = (blk >> 4) & (NH - 1);
  const int b = blk >> 8;
  const int qrow0 = qblk * 128 + w * 16;
  const int causal = *causal_flag;

  // Q fragments (head dim 64 -> two 16x32 A frags), loaded once
  FragB qf0, qf1;
  {
    const unsigned int* Q32 = (const unsigned int*)Qb;
    size_t base = ((size_t)(b * TQN + qrow0 + l16) * DM + h * HD) / 2;
#pragma unroll
    for (int v = 0; v < 8; ++v) {
      int pi = pairIdx(v, hl);
      qf0.u[v] = Q32[base + pi];
      qf1.u[v] = Q32[base + 16 + pi];
    }
  }

  v8f acc[4] = {};
  float mrow[8], lrow[8];
#pragma unroll
  for (int v = 0; v < 8; ++v) { mrow[v] = -3.0e38f; lrow[v] = 0.f; }

  const int jendBlk = causal ? (qblk * 128 + 128) : TKVN;  // block-uniform
  const int jendW = causal ? (qrow0 + 16) : TKVN;          // per-wave
  const float scale = 0.125f;  // 1/sqrt(64)

#ifdef HAVE_TDM
  // issue K (rows of 32 dwords -> pad_interval=4) and V^T (rows of 16 dwords
  // -> pad_interval=3) tile loads for a given kv offset into buffer `buf`.
  auto stage_tdm = [&](int jb, int buf) {
    unsigned long long ka = (unsigned long long)(size_t)(
        Kb + (size_t)(b * TKVN + jb) * DM + h * HD);
    tdm_load2d(ka, (unsigned)(size_t)ldsK[buf], DM, MROWS, 64, 32, DM, 4);
    unsigned long long va = (unsigned long long)(size_t)(
        Vt + (size_t)((b * NH + h) * HD) * TKVN + jb);
    tdm_load2d(va, (unsigned)(size_t)ldsV[buf], TKVN, BATCH * NH * HD, 32, 64,
               TKVN, 3);
  };
  if (w == 0) stage_tdm(0, 0);  // prologue: tile 0 in flight
#else
  const unsigned int* K32 = (const unsigned int*)Kb;
  const unsigned int* V32 = (const unsigned int*)Vt;
#endif

  int cur = 0;
  for (int jb = 0; jb < jendBlk; jb += 32, cur ^= 1) {
#ifdef HAVE_TDM
    if (w == 0) {
      if (jb + 32 < jendBlk) {
        stage_tdm(jb + 32, cur ^ 1);           // look-ahead issue
        __builtin_amdgcn_s_wait_tensorcnt(2);  // retire current tile's 2 ops
      } else {
        __builtin_amdgcn_s_wait_tensorcnt(0);
      }
    }
#else
#pragma unroll
    for (int i = 0; i < 4; ++i) {
      int e = tid + i * 256;  // 0..1023
      int kv = e >> 5, dp = e & 31;
      ldsK[cur][kv * 33 + dp] =
          K32[((size_t)(b * TKVN + jb + kv) * DM + h * HD) / 2 + dp];
      int dv = e >> 4, kp = e & 15;
      ldsV[cur][dv * 17 + kp] =
          V32[((size_t)((b * NH + h) * HD + dv) * TKVN + jb) / 2 + kp];
    }
#endif
    __syncthreads();

    if (jb < jendW) {
      // ---- S = Q K^T for 16 rows x 32 kv cols (two 16x16 tiles) ----
      v8f s0 = {}, s1 = {};
      {
        FragB ka0, ka1, kb0, kb1;
#pragma unroll
        for (int v = 0; v < 8; ++v) {
          int pi = pairIdx(v, hl);
          ka0.u[v] = ldsK[cur][l16 * 33 + pi];             // kv 0..15, d 0..31
          ka1.u[v] = ldsK[cur][l16 * 33 + 16 + pi];        // kv 0..15, d 32..63
          kb0.u[v] = ldsK[cur][(16 + l16) * 33 + pi];      // kv 16..31
          kb1.u[v] = ldsK[cur][(16 + l16) * 33 + 16 + pi];
        }
        s0 = __builtin_amdgcn_wmma_f32_16x16x32_bf16(false, qf0.v, false, ka0.v,
                                                     (short)0, s0, false, false);
        s0 = __builtin_amdgcn_wmma_f32_16x16x32_bf16(false, qf1.v, false, ka1.v,
                                                     (short)0, s0, false, false);
        s1 = __builtin_amdgcn_wmma_f32_16x16x32_bf16(false, qf0.v, false, kb0.v,
                                                     (short)0, s1, false, false);
        s1 = __builtin_amdgcn_wmma_f32_16x16x32_bf16(false, qf1.v, false, kb1.v,
                                                     (short)0, s1, false, false);
      }
      // ---- mask + scale ----
      const int jc0 = jb + l16, jc1 = jb + 16 + l16;
      const bool pad0 = padmask[b * TKVN + jc0] != 0;
      const bool pad1 = padmask[b * TKVN + jc1] != 0;
      const int qrowL = qrow0 + 8 * hl;
      float sv0[8], sv1[8];
#pragma unroll
      for (int v = 0; v < 8; ++v) {
        int q = qrowL + v;
        float x0 = s0[v] * scale;
        float x1 = s1[v] * scale;
        if (pad0 || (causal && jc0 > q)) x0 = -3.0e38f;
        if (pad1 || (causal && jc1 > q)) x1 = -3.0e38f;
        sv0[v] = x0;
        sv1[v] = x1;
      }
      // ---- online softmax (row = v + 8*hl; reduce across 16-lane group) ----
      float scl[8];
#pragma unroll
      for (int v = 0; v < 8; ++v) {
        float rm = fmaxf(sv0[v], sv1[v]);
#pragma unroll
        for (int m = 1; m < 16; m <<= 1) rm = fmaxf(rm, __shfl_xor(rm, m, 32));
        float mnew = fmaxf(mrow[v], rm);
        float sc = __expf(mrow[v] - mnew);
        float p0 = __expf(sv0[v] - mnew);
        float p1 = __expf(sv1[v] - mnew);
        float rs = p0 + p1;
#pragma unroll
        for (int m = 1; m < 16; m <<= 1) rs += __shfl_xor(rs, m, 32);
        lrow[v] = lrow[v] * sc + rs;
        mrow[v] = mnew;
        scl[v] = sc;
        sv0[v] = p0;
        sv1[v] = p1;
      }
#pragma unroll
      for (int d = 0; d < 4; ++d)
#pragma unroll
        for (int v = 0; v < 8; ++v) acc[d][v] *= scl[v];

      // ---- P: C-layout -> A-fragment layout via per-wave LDS ----
      {
        unsigned short* ps = (unsigned short*)pLds[w];
#pragma unroll
        for (int v = 0; v < 8; ++v) {
          int r = v + 8 * hl;
          ps[r * 32 + l16] = f2bf(sv0[v]);
          ps[r * 32 + 16 + l16] = f2bf(sv1[v]);
        }
      }
      asm volatile("s_wait_dscnt 0" ::: "memory");
      FragB pf;
#pragma unroll
      for (int v = 0; v < 8; ++v) pf.u[v] = pLds[w][l16 * 16 + pairIdx(v, hl)];

      // ---- acc += P @ V (V^T tile in LDS, pairs along kv contiguous) ----
#pragma unroll
      for (int d = 0; d < 4; ++d) {
        FragB vf;
#pragma unroll
        for (int v = 0; v < 8; ++v)
          vf.u[v] = ldsV[cur][(d * 16 + l16) * 17 + pairIdx(v, hl)];
        acc[d] = __builtin_amdgcn_wmma_f32_16x16x32_bf16(
            false, pf.v, false, vf.v, (short)0, acc[d], false, false);
      }
    }
    __syncthreads();
  }

  // ---- normalize + store bf16 (B*TQ, H*DV) ----
#pragma unroll
  for (int d = 0; d < 4; ++d) {
#pragma unroll
    for (int v = 0; v < 8; ++v) {
      int row = qrow0 + v + 8 * hl;
      float o = acc[d][v] / lrow[v];
      Ob[(size_t)(b * TQN + row) * DM + h * HD + d * 16 + l16] = f2bf(o);
    }
  }
}

// ---------------------------------------------------------------------------
extern "C" void kernel_launch(void* const* d_in, const int* in_sizes, int n_in,
                              void* d_out, int out_size, void* d_ws,
                              size_t ws_size, hipStream_t stream) {
  const float* q_seq = (const float*)d_in[0];
  const float* kv_seq = (const float*)d_in[1];
  const unsigned char* padmask = (const unsigned char*)d_in[2];
  const float* Wq = (const float*)d_in[3];
  const float* bq = (const float*)d_in[4];
  const float* Wkv = (const float*)d_in[5];
  const float* bkv = (const float*)d_in[6];
  const float* Wo = (const float*)d_in[7];
  const float* bo = (const float*)d_in[8];
  const int* causal = (const int*)d_in[9];

  unsigned short* wsQ = (unsigned short*)d_ws;                    // 16 MB
  unsigned short* wsK = wsQ + (size_t)MROWS * DM;                 // 16 MB
  unsigned short* wsVt = wsK + (size_t)MROWS * DM;                // 16 MB
  unsigned short* wsA = wsVt + (size_t)MROWS * DM;                // 16 MB

  dim3 blk(256);

  // Q projection: (8192,1024) x (1024,1024) -> bf16 Q
  mha_gemm_kernel<false, 0><<<dim3(DM / 64, MROWS / 128), blk, 0, stream>>>(
      q_seq, Wq, bq, wsQ, nullptr, MROWS, DM, DM);

  // KV projection: (8192,1024) x (1024,2048) -> bf16 K (row-major) + V^T
  mha_gemm_kernel<false, 1><<<dim3(2 * DM / 64, MROWS / 128), blk, 0, stream>>>(
      kv_seq, Wkv, bkv, wsK, wsVt, MROWS, DM, 2 * DM);

  // Flash attention -> bf16 attn-out (8192,1024)
  mha_flash_kernel<<<dim3(BATCH * NH * (TQN / 128)), blk, 0, stream>>>(
      wsQ, wsK, wsVt, padmask, causal, wsA);

  // Output projection: (8192,1024)bf16 x (1024,1024) -> f32 d_out
  mha_gemm_kernel<true, 2><<<dim3(DM / 64, MROWS / 128), blk, 0, stream>>>(
      wsA, Wo, bo, d_out, nullptr, MROWS, DM, DM);
}
